// LSTM_37417755082919
// MI455X (gfx1250) — matrix operand order repaired and checked by reading, not compiled, not run on hardware
//
#include <hip/hip_runtime.h>
#include <hip/hip_bf16.h>
#include <stdint.h>

// LSTM: S=512, B=64, I=H=1024.
// bf16 WMMA (16x16x32, f32 acc). One wave computes one 16x16 output tile for
// ALL FOUR gates (x-projection and h-projection folded into one K=2048
// reduction), then applies the cell nonlinearity locally. 512 sequential step
// kernels; h state ping-pongs in bf16, c state stays fp32 in workspace.
// This revision software-pipelines the K-loop (prefetch next k-tile's A + 4 B
// fragments into distinct registers) so the compiler can overlap the ten
// global_load_b128 of iteration k+1 with the four WMMAs of iteration k instead
// of draining loadcnt to 0 before every WMMA.

typedef __bf16 bf16_t;
typedef __attribute__((ext_vector_type(8)))  __bf16 v8bf;
typedef __attribute__((ext_vector_type(16))) __bf16 v16bf;
typedef __attribute__((ext_vector_type(8)))  float  v8f;

#define S_LEN 512
#define BATCH 64
#define IDIM  1024
#define HDIM  1024

// ---------------- fragment loads ----------------
__device__ __forceinline__ v16bf cat8(v8bf lo, v8bf hi) {
  return __builtin_shufflevector(lo, hi, 0,1,2,3,4,5,6,7,8,9,10,11,12,13,14,15);
}
// 16-bit A (16x32): lane<16 holds row M=lane, K = 0..15; lane>=16 row
// M=lane-16, K = 16..31. Caller pre-offsets pointer by hs*16.
__device__ __forceinline__ v16bf load_a_frag(const bf16_t* p) {
  v8bf lo = *(const v8bf*)(p);        // K base + 0..7
  v8bf hi = *(const v8bf*)(p + 8);    // K base + 8..15
  return cat8(lo, hi);
}
// 16-bit B (32x16): lane<16 = column N=lane, K=0..15; lane>=16 column N=lane-16,
// K=16..31 (contiguous k order chosen here, same as A). Column n of B is row n of W
// (z = h @ W^T), contiguous in memory. Caller pre-offsets pointer by hs*16.
__device__ __forceinline__ v16bf load_b_frag(const bf16_t* p) {
  v8bf lo = *(const v8bf*)(p);        // K base + 0..7
  v8bf hi = *(const v8bf*)(p + 8);    // K base + 8..15
  return cat8(lo, hi);
}

__device__ __forceinline__ float fast_sigmoid(float x) {
  return 1.0f / (1.0f + __expf(-x));
}
__device__ __forceinline__ float fast_tanh(float x) {
  return 2.0f / (1.0f + __expf(-2.0f * x)) - 1.0f;
}

__device__ __forceinline__ v8f wmma_bf16(v16bf a, v16bf b, v8f c) {
  return __builtin_amdgcn_wmma_f32_16x16x32_bf16(false, a, false, b, (short)0,
                                                 c, false, false);
}

// ---------------- prep kernels ----------------
__global__ void cvt_f32_bf16(const float* __restrict__ src,
                             bf16_t* __restrict__ dst, int n) {
  int i = blockIdx.x * blockDim.x + threadIdx.x;
  if (i < n) dst[i] = (bf16_t)src[i];
}

__global__ void bias_combine(const float* __restrict__ a,
                             const float* __restrict__ b,
                             const float* __restrict__ c,
                             float* __restrict__ d, int n) {
  int i = blockIdx.x * blockDim.x + threadIdx.x;
  if (i < n) d[i] = a[i] + b[i] + c[i];
}

__global__ void zero_u32(uint32_t* __restrict__ p, int n) {
  int i = blockIdx.x * blockDim.x + threadIdx.x;
  if (i < n) p[i] = 0u;
}

// ------- GEMM phase: 4 gates, K = 32*ktiles, 1-deep software pipeline -------
__device__ __forceinline__ void gemm_phase(const bf16_t* __restrict__ aRow,
                                           const bf16_t* __restrict__ bRow,
                                           int ktiles,
                                           v8f& acc0, v8f& acc1,
                                           v8f& acc2, v8f& acc3) {
  const size_t GS = (size_t)HDIM * IDIM;  // per-gate weight stride

  // prologue: fragments for k-tile 0
  v16bf a  = load_a_frag(aRow);
  v16bf b0 = load_b_frag(bRow);
  v16bf b1 = load_b_frag(bRow + GS);
  v16bf b2 = load_b_frag(bRow + 2 * GS);
  v16bf b3 = load_b_frag(bRow + 3 * GS);

  for (int kt = 0; kt < ktiles - 1; ++kt) {
    const bf16_t* aN = aRow + (size_t)(kt + 1) * 32;
    const bf16_t* bN = bRow + (size_t)(kt + 1) * 32;
    // prefetch next k-tile into distinct registers (stays in flight across
    // the WMMAs below -> nonzero s_wait_loadcnt instead of full drain)
    v16bf a2 = load_a_frag(aN);
    v16bf c0 = load_b_frag(bN);
    v16bf c1 = load_b_frag(bN + GS);
    v16bf c2 = load_b_frag(bN + 2 * GS);
    v16bf c3 = load_b_frag(bN + 3 * GS);

    acc0 = wmma_bf16(a, b0, acc0);
    acc1 = wmma_bf16(a, b1, acc1);
    acc2 = wmma_bf16(a, b2, acc2);
    acc3 = wmma_bf16(a, b3, acc3);

    a = a2; b0 = c0; b1 = c1; b2 = c2; b3 = c3;
  }
  // epilogue: last k-tile
  acc0 = wmma_bf16(a, b0, acc0);
  acc1 = wmma_bf16(a, b1, acc1);
  acc2 = wmma_bf16(a, b2, acc2);
  acc3 = wmma_bf16(a, b3, acc3);
}

// ---------------- one LSTM step ----------------
// Grid: 32 blocks x 256 threads = 256 wave32 waves. Wave w: m_tile = w&3 (16
// batch rows), n_tile = w>>2 (16 hidden cols). Each wave: K=2048 reduction,
// 4 gates -> 256 WMMAs, then cell update.
__global__ __launch_bounds__(256, 1)
void lstm_step(const bf16_t* __restrict__ xt,     // [64][1024] bf16 (step t)
               const bf16_t* __restrict__ h_in,   // [64][1024] bf16
               bf16_t* __restrict__ h_out,        // [64][1024] bf16
               const bf16_t* __restrict__ wx,     // [4][1024][1024] bf16
               const bf16_t* __restrict__ wh,     // [4][1024][1024] bf16
               const float*  __restrict__ bias,   // [4][1024] combined
               float* __restrict__ c_state,       // [64][1024] f32
               float* __restrict__ out_t,         // d_out + t*B*H
               float* __restrict__ h_fin,         // d_out final-h slot
               float* __restrict__ c_fin) {       // d_out final-c slot
  const int lane   = threadIdx.x & 31;
  const int wave   = blockIdx.x * (blockDim.x >> 5) + (threadIdx.x >> 5);
  const int m_tile = wave & 3;
  const int n_tile = wave >> 2;
  const int lane16 = lane & 15;
  const int hs     = lane >> 4;
  const int col    = n_tile * 16 + lane16;

  const bf16_t* aX = xt   + (size_t)(m_tile * 16 + lane16) * IDIM + hs * 16;
  const bf16_t* aH = h_in + (size_t)(m_tile * 16 + lane16) * HDIM + hs * 16;
  const bf16_t* bX = wx + (size_t)col * IDIM + hs * 16;
  const bf16_t* bH = wh + (size_t)col * HDIM + hs * 16;

  v8f acc0 = {}, acc1 = {}, acc2 = {}, acc3 = {};
  gemm_phase(aX, bX, IDIM / 32, acc0, acc1, acc2, acc3);  // x @ Wx^T
  gemm_phase(aH, bH, HDIM / 32, acc0, acc1, acc2, acc3);  // h @ Wh^T

  const float bi = bias[0 * HDIM + col];
  const float bf = bias[1 * HDIM + col];
  const float bc = bias[2 * HDIM + col];
  const float bo = bias[3 * HDIM + col];
  const int row0 = m_tile * 16 + hs * 8;  // C/D layout: VGPR e -> M = e (+8 hi half)
#pragma unroll
  for (int e = 0; e < 8; ++e) {
    const int idx = (row0 + e) * HDIM + col;
    const float ig = fast_sigmoid(acc0[e] + bi);
    const float fg = fast_sigmoid(acc1[e] + bf);
    const float gg = fast_tanh   (acc2[e] + bc);
    const float og = fast_sigmoid(acc3[e] + bo);
    const float cn = fg * c_state[idx] + ig * gg;
    const float hn = og * fast_tanh(cn);
    c_state[idx] = cn;
    out_t[idx]   = hn;
    h_out[idx]   = (bf16_t)hn;
    h_fin[idx]   = hn;   // last step's values persist in d_out
    c_fin[idx]   = cn;
  }
}

// ---------------- host ----------------
extern "C" void kernel_launch(void* const* d_in, const int* in_sizes, int n_in,
                              void* d_out, int out_size, void* d_ws, size_t ws_size,
                              hipStream_t stream) {
  (void)in_sizes; (void)n_in; (void)out_size; (void)ws_size;
  uint8_t* ws = (uint8_t*)d_ws;

  // workspace layout (bytes)
  bf16_t*  x_bf = (bf16_t*)(ws);                    // 512*64*1024*2 = 64 MB
  bf16_t*  wx   = (bf16_t*)(ws + 67108864ULL);      // 4*1024*1024*2 = 8 MB
  bf16_t*  wh   = (bf16_t*)(ws + 75497472ULL);      // 8 MB
  float*   bias = (float*) (ws + 83886080ULL);      // 4*1024*4 = 16 KB
  bf16_t*  hbuf = (bf16_t*)(ws + 83902464ULL);      // 2*64*1024*2 = 256 KB (ping-pong)
  float*   cst  = (float*) (ws + 84164608ULL);      // 64*1024*4 = 256 KB

  // convert x to bf16
  {
    const int n = S_LEN * BATCH * IDIM;
    cvt_f32_bf16<<<(n + 255) / 256, 256, 0, stream>>>((const float*)d_in[0], x_bf, n);
  }
  // per-gate inputs: w_x = d_in[1+5g], b_x = d_in[2+5g], w_h = d_in[3+5g],
  //                  b_h = d_in[4+5g], b = d_in[5+5g]
  for (int g = 0; g < 4; ++g) {
    const int n = HDIM * IDIM;
    cvt_f32_bf16<<<(n + 255) / 256, 256, 0, stream>>>(
        (const float*)d_in[1 + 5 * g], wx + (size_t)g * n, n);
    cvt_f32_bf16<<<(n + 255) / 256, 256, 0, stream>>>(
        (const float*)d_in[3 + 5 * g], wh + (size_t)g * n, n);
    bias_combine<<<(HDIM + 255) / 256, 256, 0, stream>>>(
        (const float*)d_in[2 + 5 * g], (const float*)d_in[4 + 5 * g],
        (const float*)d_in[5 + 5 * g], bias + (size_t)g * HDIM, HDIM);
  }
  // zero h ping-pong (bf16) + c state (f32): contiguous 524288 B = 131072 words
  zero_u32<<<(131072 + 255) / 256, 256, 0, stream>>>((uint32_t*)hbuf, 131072);

  float* out   = (float*)d_out;
  float* h_fin = out + (size_t)S_LEN * BATCH * HDIM;
  float* c_fin = h_fin + (size_t)BATCH * HDIM;

  for (int t = 0; t < S_LEN; ++t) {
    const bf16_t* xt   = x_bf + (size_t)t * BATCH * IDIM;
    const bf16_t* hin  = hbuf + (size_t)(t & 1) * (BATCH * HDIM);
    bf16_t*       hout = hbuf + (size_t)((t + 1) & 1) * (BATCH * HDIM);
    lstm_step<<<32, 256, 0, stream>>>(xt, hin, hout, wx, wh, bias, cst,
                                      out + (size_t)t * BATCH * HDIM,
                                      h_fin, c_fin);
  }
}
